// NonLocal_42580305773252
// MI455X (gfx1250) — matrix operand order repaired and checked by reading, not compiled, not run on hardware
//
#include <hip/hip_runtime.h>
#include <hip/hip_bf16.h>
#include <math.h>

// ---------------- problem constants ----------------
constexpr int B_  = 8;
constexpr int C_  = 512;
constexpr int H_  = 64;
constexpr int W_  = 64;
constexpr int CI_ = 256;
constexpr int N_  = H_ * W_;          // 4096
constexpr int M_  = (H_/2) * (W_/2);  // 1024

// ---------------- workspace layout (bytes) ----------------
constexpr size_t OFF_XT   = 0;                                   // f16 [B][N][C]   32 MB
constexpr size_t OFF_TH   = OFF_XT  + (size_t)B_*N_*C_*2;        // f16 [B][N][CI]  16 MB
constexpr size_t OFF_G    = OFF_TH  + (size_t)B_*N_*CI_*2;       // f16 [B][N][CI]  16 MB (reused as yT)
constexpr size_t OFF_PH   = OFF_G   + (size_t)B_*N_*CI_*2;       // f16 [B][N][CI]  16 MB
constexpr size_t OFF_PHP  = OFF_PH  + (size_t)B_*N_*CI_*2;       // f16 [B][M][CI]   4 MB
constexpr size_t OFF_GCM  = OFF_PHP + (size_t)B_*M_*CI_*2;       // f16 [B][CI][M]   4 MB
constexpr size_t OFF_WG16 = OFF_GCM + (size_t)B_*CI_*M_*2;       // f16 [CI][C]
constexpr size_t OFF_WT16 = OFF_WG16 + (size_t)CI_*C_*2;
constexpr size_t OFF_WP16 = OFF_WT16 + (size_t)CI_*C_*2;
constexpr size_t OFF_WO16 = OFF_WP16 + (size_t)CI_*C_*2;         // f16 [C][CI]
constexpr size_t OFF_SA   = OFF_WO16 + (size_t)C_*CI_*2;         // f32 [C]
constexpr size_t OFF_SB   = OFF_SA + (size_t)C_*4;               // f32 [C]

// ---------------- WMMA types/helpers ----------------
typedef _Float16 v16h __attribute__((ext_vector_type(16)));
typedef _Float16 v8h  __attribute__((ext_vector_type(8)));
typedef float    v8f  __attribute__((ext_vector_type(8)));

#define WMMA_F16(A, Bm, Cm) \
  __builtin_amdgcn_wmma_f32_16x16x32_f16(false, (A), false, (Bm), (short)0, (Cm), false, false)

// A operand (16 rows x 32 K, f16) from row-major [row][k] (ld = elements).
// Lane L<16: row=row0+L, K {k0+0..7, k0+16..23}; lane>=16: row=row0+L-16, K {k0+8..15, k0+24..31}.
__device__ __forceinline__ v16h load_a16(const _Float16* base, int ld, int row0, int k0, int lane) {
  const int r  = row0 + (lane & 15);
  const int kk = k0 + ((lane & 16) ? 8 : 0);
  const _Float16* p = base + (size_t)r * ld + kk;
  v8h lo = *(const v8h*)(p);
  v8h hi = *(const v8h*)(p + 16);
  v16h o;
#pragma unroll
  for (int i = 0; i < 8; ++i) { o[i] = lo[i]; o[i + 8] = hi[i]; }
  return o;
}

// B operand (32 K x 16 cols, f16) from K-contiguous [col][k] (ld = elements).
// Lane holds col = col0 + (L&15); lane<16: K {k0+0..7, k0+16..23}; lane>=16: K {k0+8..15, k0+24..31}.
__device__ __forceinline__ v16h load_b16(const _Float16* base, int ld, int col0, int k0, int lane) {
  const int c  = col0 + (lane & 15);
  const int kk = k0 + ((lane & 16) ? 8 : 0);
  const _Float16* p = base + (size_t)c * ld + kk;
  v8h lo = *(const v8h*)(p);
  v8h hi = *(const v8h*)(p + 16);
  v16h o;
#pragma unroll
  for (int i = 0; i < 8; ++i) { o[i] = lo[i]; o[i + 8] = hi[i]; }
  return o;
}

// ---------------- small prep kernels ----------------
__global__ void k_convert_f16(const float* __restrict__ src, _Float16* __restrict__ dst, int n) {
  int i = blockIdx.x * blockDim.x + threadIdx.x;
  if (i < n) dst[i] = (_Float16)src[i];
}

__global__ void k_bn_prep(const float* __restrict__ b_out, const float* __restrict__ gamma,
                          const float* __restrict__ beta, const float* __restrict__ mean,
                          const float* __restrict__ var, float* __restrict__ sA,
                          float* __restrict__ sB) {
  int i = blockIdx.x * blockDim.x + threadIdx.x;
  if (i < C_) {
    float inv = gamma[i] * rsqrtf(var[i] + 1e-5f);
    sA[i] = inv;
    sB[i] = (b_out[i] - mean[i]) * inv + beta[i];
  }
}

// xT[b][n][c] (f16) = x[b][c][n] (f32)
__global__ void k_transpose_x(const float* __restrict__ x, _Float16* __restrict__ xT) {
  int i = blockIdx.x * blockDim.x + threadIdx.x;   // over B*N*C = 2^24
  int c = i & (C_ - 1);
  int n = (i >> 9) & (N_ - 1);
  int b = i >> 21;
  xT[i] = (_Float16)x[((size_t)b * C_ + c) * N_ + n];
}

// ---------------- conv GEMM: outT[b][n][o] = sum_c xT[b][n][c]*W[o][c] + bias[o] ----------------
// 16x64 tile per wave, 4 accumulators: low register pressure -> double-buffered loads.
__global__ void k_gemm_conv(const _Float16* __restrict__ xT, const _Float16* __restrict__ Wt,
                            const float* __restrict__ bias, _Float16* __restrict__ outT) {
  const int lane = threadIdx.x & 31;
  const int wg   = blockIdx.x * (blockDim.x >> 5) + (threadIdx.x >> 5);
  const int ci_blk = wg & 3;            // CI/64 = 4
  const int n_blk  = (wg >> 2) & 255;   // N/16 = 256
  const int b      = wg >> 10;
  const int n0 = n_blk * 16, o0 = ci_blk * 64;
  const _Float16* xb = xT + (size_t)b * N_ * C_;

  v8f acc[4] = {};
  for (int k = 0; k < C_; k += 32) {
    v16h a = load_a16(xb, C_, n0, k, lane);
#pragma unroll
    for (int t = 0; t < 4; ++t) {
      v16h bm = load_b16(Wt, C_, o0 + t * 16, k, lane);
      acc[t] = WMMA_F16(a, bm, acc[t]);
    }
  }
  const int col = lane & 15, hi8 = (lane & 16) ? 8 : 0;
  _Float16* ob = outT + (size_t)b * N_ * CI_;
#pragma unroll
  for (int t = 0; t < 4; ++t) {
    float bv = bias[o0 + t * 16 + col];
#pragma unroll
    for (int r = 0; r < 8; ++r)
      ob[(size_t)(n0 + r + hi8) * CI_ + o0 + t * 16 + col] = (_Float16)(acc[t][r] + bv);
  }
}

// ---------------- 2x2 maxpool variants ----------------
// dst[b][m][ci] = max over 2x2 of src[b][n][ci]
__global__ void k_pool_nc(const _Float16* __restrict__ src, _Float16* __restrict__ dst) {
  int i = blockIdx.x * blockDim.x + threadIdx.x;   // B*M*CI = 2^21
  int ci = i & (CI_ - 1);
  int m  = (i >> 8) & (M_ - 1);
  int b  = i >> 18;
  int mh = m >> 5, mw = m & 31;
  size_t base = ((size_t)b * N_ + (2 * mh) * 64 + 2 * mw) * CI_ + ci;
  float v0 = (float)src[base];
  float v1 = (float)src[base + CI_];
  float v2 = (float)src[base + (size_t)64 * CI_];
  float v3 = (float)src[base + (size_t)65 * CI_];
  dst[i] = (_Float16)fmaxf(fmaxf(v0, v1), fmaxf(v2, v3));
}

// dst[b][ci][m] = max over 2x2 of src[b][n][ci]  (transposed for B-operand loads)
__global__ void k_pool_cm(const _Float16* __restrict__ src, _Float16* __restrict__ dst) {
  int i = blockIdx.x * blockDim.x + threadIdx.x;   // B*CI*M = 2^21
  int m  = i & (M_ - 1);
  int ci = (i >> 10) & (CI_ - 1);
  int b  = i >> 18;
  int mh = m >> 5, mw = m & 31;
  size_t base = ((size_t)b * N_ + (2 * mh) * 64 + 2 * mw) * CI_ + ci;
  float v0 = (float)src[base];
  float v1 = (float)src[base + CI_];
  float v2 = (float)src[base + (size_t)64 * CI_];
  float v3 = (float)src[base + (size_t)65 * CI_];
  dst[i] = (_Float16)fmaxf(fmaxf(v0, v1), fmaxf(v2, v3));
}

// ---------------- fused attention: scores -> softmax (in place) -> y ----------------
// One wave per 16 query rows. 32 KB LDS holds the 16x1024 score/prob tile.
__global__ void __launch_bounds__(32) k_attn(const _Float16* __restrict__ thetaT,
                                             const _Float16* __restrict__ phiP,
                                             const _Float16* __restrict__ gCM,
                                             _Float16* __restrict__ yT) {
  extern __shared__ _Float16 sS[];   // 16 x 1024 f16 (32 KB)
  const int lane = threadIdx.x;
  const int b  = blockIdx.x >> 8;
  const int n0 = (blockIdx.x & 255) * 16;
  const int col = lane & 15, hi8 = (lane & 16) ? 8 : 0;

  const _Float16* th = thetaT + (size_t)b * N_ * CI_;
  const _Float16* ph = phiP   + (size_t)b * M_ * CI_;

  // Preload the 16x256 theta block as A operands (held in registers across all m-tiles).
  v16h areg[8];
#pragma unroll
  for (int kk = 0; kk < 8; ++kk) areg[kk] = load_a16(th, CI_, n0, kk * 32, lane);

  // Phase 1: S[16][1024] = theta_rows @ phi^T   (K = CI = 256)
  for (int mt = 0; mt < M_ / 16; ++mt) {
    v8f acc = {};
#pragma unroll
    for (int kk = 0; kk < 8; ++kk) {
      v16h bm = load_b16(ph, CI_, mt * 16, kk * 32, lane);
      acc = WMMA_F16(areg[kk], bm, acc);
    }
#pragma unroll
    for (int r = 0; r < 8; ++r)
      sS[(r + hi8) * M_ + mt * 16 + col] = (_Float16)acc[r];
  }
  __syncthreads();

  // Phase 2: softmax over m, in place. Row = lane&15; the two half-rows pair via shfl_xor(16).
  {
    const int row  = lane & 15;
    const int half = lane >> 4;
    v8h* rp = (v8h*)(sS + row * M_ + half * 512);
    float mx = -3.0e38f;
    for (int j = 0; j < 64; ++j) {
      v8h v = rp[j];
#pragma unroll
      for (int e = 0; e < 8; ++e) mx = fmaxf(mx, (float)v[e]);
    }
    mx = fmaxf(mx, __shfl_xor(mx, 16, 32));
    float sum = 0.0f;
    for (int j = 0; j < 64; ++j) {
      v8h v = rp[j];
      v8h o;
#pragma unroll
      for (int e = 0; e < 8; ++e) {
        float ex = __expf((float)v[e] - mx);
        sum += ex;
        o[e] = (_Float16)ex;
      }
      rp[j] = o;
    }
    sum += __shfl_xor(sum, 16, 32);
    float inv = 1.0f / sum;
    for (int j = 0; j < 64; ++j) {
      v8h v = rp[j];
#pragma unroll
      for (int e = 0; e < 8; ++e) v[e] = (_Float16)((float)v[e] * inv);
      rp[j] = v;
    }
  }
  __syncthreads();

  // Phase 3: y[16][CI] = P @ g^T  (K = M = 1024).
  // Two passes of 8 ci-tiles (8 accumulators each) keep VGPR pressure low enough
  // for the scheduler to double-buffer the B loads between WMMAs.
  const _Float16* gb = gCM + (size_t)b * CI_ * M_;
  _Float16*       yb = yT  + (size_t)b * N_ * CI_;
#pragma unroll
  for (int hf = 0; hf < 2; ++hf) {
    v8f acc[8];
#pragma unroll
    for (int ct = 0; ct < 8; ++ct) acc[ct] = v8f{};
    for (int k = 0; k < M_; k += 32) {
      v16h a = load_a16(sS, M_, 0, k, lane);
#pragma unroll
      for (int ct = 0; ct < 8; ++ct) {
        v16h bm = load_b16(gb, M_, (hf * 8 + ct) * 16, k, lane);
        acc[ct] = WMMA_F16(a, bm, acc[ct]);
      }
    }
#pragma unroll
    for (int ct = 0; ct < 8; ++ct)
#pragma unroll
      for (int r = 0; r < 8; ++r)
        yb[(size_t)(n0 + r + hi8) * CI_ + (hf * 8 + ct) * 16 + col] = (_Float16)acc[ct][r];
  }
}

// ---------------- output conv + BN + residual: out[b][co][n] (f32), 32x64 tiles ----------------
__global__ void k_gemm_out(const _Float16* __restrict__ Wo, const _Float16* __restrict__ yT,
                           const float* __restrict__ sA, const float* __restrict__ sB,
                           const float* __restrict__ x, float* __restrict__ out) {
  const int lane = threadIdx.x & 31;
  const int wg   = blockIdx.x * (blockDim.x >> 5) + (threadIdx.x >> 5);
  const int n_blk  = wg & 63;          // N/64
  const int co_blk = (wg >> 6) & 15;   // C/32
  const int b      = wg >> 10;
  const int n0 = n_blk * 64, co0 = co_blk * 32;
  const _Float16* yb = yT + (size_t)b * N_ * CI_;

  v8f acc[2][4] = {};
  for (int k = 0; k < CI_; k += 32) {
    v16h a0 = load_a16(Wo, CI_, co0,      k, lane);
    v16h a1 = load_a16(Wo, CI_, co0 + 16, k, lane);
#pragma unroll
    for (int t = 0; t < 4; ++t) {
      v16h bm = load_b16(yb, CI_, n0 + t * 16, k, lane);
      acc[0][t] = WMMA_F16(a0, bm, acc[0][t]);
      acc[1][t] = WMMA_F16(a1, bm, acc[1][t]);
    }
  }
  const int col = lane & 15, hi8 = (lane & 16) ? 8 : 0;
#pragma unroll
  for (int s = 0; s < 2; ++s) {
#pragma unroll
    for (int r = 0; r < 8; ++r) {
      int co = co0 + s * 16 + r + hi8;
      float a_ = sA[co], b_ = sB[co];
      size_t rowbase = ((size_t)b * C_ + co) * N_;
#pragma unroll
      for (int t = 0; t < 4; ++t) {
        size_t idx = rowbase + n0 + t * 16 + col;
        out[idx] = acc[s][t][r] * a_ + b_ + x[idx];
      }
    }
  }
}

// ---------------- launch ----------------
extern "C" void kernel_launch(void* const* d_in, const int* in_sizes, int n_in,
                              void* d_out, int out_size, void* d_ws, size_t ws_size,
                              hipStream_t stream) {
  const float* x    = (const float*)d_in[0];
  const float* w_g  = (const float*)d_in[1];
  const float* b_g  = (const float*)d_in[2];
  const float* w_t  = (const float*)d_in[3];
  const float* b_t  = (const float*)d_in[4];
  const float* w_p  = (const float*)d_in[5];
  const float* b_p  = (const float*)d_in[6];
  const float* w_o  = (const float*)d_in[7];
  const float* b_o  = (const float*)d_in[8];
  const float* gmma = (const float*)d_in[9];
  const float* beta = (const float*)d_in[10];
  const float* mean = (const float*)d_in[11];
  const float* var  = (const float*)d_in[12];
  float* out = (float*)d_out;

  char* ws = (char*)d_ws;
  _Float16* xT     = (_Float16*)(ws + OFF_XT);
  _Float16* thetaT = (_Float16*)(ws + OFF_TH);
  _Float16* gT     = (_Float16*)(ws + OFF_G);    // conv-g output, later reused as yT
  _Float16* phiT   = (_Float16*)(ws + OFF_PH);
  _Float16* phiP   = (_Float16*)(ws + OFF_PHP);
  _Float16* gCM    = (_Float16*)(ws + OFF_GCM);
  _Float16* wg16   = (_Float16*)(ws + OFF_WG16);
  _Float16* wt16   = (_Float16*)(ws + OFF_WT16);
  _Float16* wp16   = (_Float16*)(ws + OFF_WP16);
  _Float16* wo16   = (_Float16*)(ws + OFF_WO16);
  float*    sA     = (float*)(ws + OFF_SA);
  float*    sB     = (float*)(ws + OFF_SB);

  // weight conversion + BN folding
  k_convert_f16<<<(CI_ * C_ + 255) / 256, 256, 0, stream>>>(w_g, wg16, CI_ * C_);
  k_convert_f16<<<(CI_ * C_ + 255) / 256, 256, 0, stream>>>(w_t, wt16, CI_ * C_);
  k_convert_f16<<<(CI_ * C_ + 255) / 256, 256, 0, stream>>>(w_p, wp16, CI_ * C_);
  k_convert_f16<<<(C_ * CI_ + 255) / 256, 256, 0, stream>>>(w_o, wo16, C_ * CI_);
  k_bn_prep<<<2, 256, 0, stream>>>(b_o, gmma, beta, mean, var, sA, sB);

  // x -> f16 [b][n][c]
  k_transpose_x<<<(B_ * N_ * C_) / 256, 256, 0, stream>>>(x, xT);

  // three 1x1 convs as WMMA GEMMs (outputs in [n][o] layout); xT stays L2-resident
  k_gemm_conv<<<2048, 128, 0, stream>>>(xT, wt16, b_t, thetaT);
  k_gemm_conv<<<2048, 128, 0, stream>>>(xT, wg16, b_g, gT);
  k_gemm_conv<<<2048, 128, 0, stream>>>(xT, wp16, b_p, phiT);

  // 2x2 maxpool into attention-friendly layouts
  k_pool_nc<<<(B_ * M_ * CI_) / 256, 256, 0, stream>>>(phiT, phiP);
  k_pool_cm<<<(B_ * CI_ * M_) / 256, 256, 0, stream>>>(gT, gCM);

  // fused scores -> softmax -> y   (yT reuses the gT buffer; g already consumed into gCM)
  _Float16* yT = gT;
  k_attn<<<B_ * (N_ / 16), 32, 32 * 1024, stream>>>(thetaT, phiP, gCM, yT);

  // output conv + BN + residual
  k_gemm_out<<<2048, 128, 0, stream>>>(wo16, yT, sA, sB, x, out);

  (void)in_sizes; (void)n_in; (void)out_size; (void)ws_size;
}
